// ModernGraphTransformer_33011118637079
// MI455X (gfx1250) — matrix operand order repaired and checked by reading, not compiled, not run on hardware
//
#include <hip/hip_runtime.h>
#include <hip/hip_bf16.h>

// ---------------- problem constants ----------------
#define NN   20000
#define EE   320000
#define DD   256
#define HH   8
#define DHH  32
#define EDD  64
#define FFF  1024

typedef __attribute__((ext_vector_type(16))) __bf16 v16bf;
typedef __attribute__((ext_vector_type(8)))  float  v8f;

// ---------------- helpers ----------------
__device__ inline unsigned fkey(float f) {           // monotone float->uint map
    unsigned u = __float_as_uint(f);
    return (u & 0x80000000u) ? ~u : (u | 0x80000000u);
}
__device__ inline float fkey_dec(unsigned k) {
    unsigned u = (k & 0x80000000u) ? (k & 0x7fffffffu) : ~k;
    return __uint_as_float(u);
}
__device__ inline __hip_bfloat16 to_bf(float f)          { return __float2bfloat16(f); }
__device__ inline __hip_bfloat16 to_bf(__hip_bfloat16 h) { return h; }

// ---------------- LayerNorm: one wave per 256-float row ----------------
__global__ __launch_bounds__(256) void ln_kernel(const float* __restrict__ x,
                                                 const float* __restrict__ g,
                                                 const float* __restrict__ b,
                                                 float* __restrict__ out, int nrows) {
    int w    = (int)((blockIdx.x * blockDim.x + threadIdx.x) >> 5);
    int lane = threadIdx.x & 31;
    if (w >= nrows) return;
    const float4* xr = (const float4*)(x + (size_t)w * DD);
    float4 a0 = xr[lane];
    float4 a1 = xr[lane + 32];
    float s  = a0.x + a0.y + a0.z + a0.w + a1.x + a1.y + a1.z + a1.w;
    float ss = a0.x*a0.x + a0.y*a0.y + a0.z*a0.z + a0.w*a0.w
             + a1.x*a1.x + a1.y*a1.y + a1.z*a1.z + a1.w*a1.w;
    #pragma unroll
    for (int o = 16; o > 0; o >>= 1) { s += __shfl_xor(s, o, 32); ss += __shfl_xor(ss, o, 32); }
    float mean = s * (1.0f / DD);
    float var  = ss * (1.0f / DD) - mean * mean;
    float r    = rsqrtf(var + 1e-5f);
    const float4* g4 = (const float4*)g;
    const float4* b4 = (const float4*)b;
    float4* o4 = (float4*)(out + (size_t)w * DD);
    float4 gg = g4[lane], bb = b4[lane], r0;
    r0.x = (a0.x - mean) * r * gg.x + bb.x;
    r0.y = (a0.y - mean) * r * gg.y + bb.y;
    r0.z = (a0.z - mean) * r * gg.z + bb.z;
    r0.w = (a0.w - mean) * r * gg.w + bb.w;
    o4[lane] = r0;
    gg = g4[lane + 32]; bb = b4[lane + 32];
    float4 r1;
    r1.x = (a1.x - mean) * r * gg.x + bb.x;
    r1.y = (a1.y - mean) * r * gg.y + bb.y;
    r1.z = (a1.z - mean) * r * gg.z + bb.z;
    r1.w = (a1.w - mean) * r * gg.w + bb.w;
    o4[lane + 32] = r1;
}

// ---------------- operand packers (WMMA per-lane layouts, ISA 7.12.2) ----------------
// A tile 16(M)x32(K) bf16: lane<16 -> row=lane, elems = K[0..7],K[16..23]
//                          lane>=16 -> row=lane-16, elems = K[8..15],K[24..31]
template <typename T>
__global__ __launch_bounds__(256) void packA_kernel(const T* __restrict__ src,
                                                    __hip_bfloat16* __restrict__ dst,
                                                    int M, int K) {
    size_t idx = (size_t)blockIdx.x * blockDim.x + threadIdx.x;
    size_t tot = (size_t)M * K;
    if (idx >= tot) return;
    int m = (int)(idx / K), k = (int)(idx % K);
    int mb = m >> 4, kb = k >> 5, sub = k & 31;
    int lane, elem;
    if      (sub <  8) { lane =  m & 15;        elem = sub;       }
    else if (sub < 16) { lane = (m & 15) + 16;  elem = sub - 8;   }
    else if (sub < 24) { lane =  m & 15;        elem = sub - 8;   }
    else               { lane = (m & 15) + 16;  elem = sub - 16;  }
    int KB = K >> 5;
    dst[(((size_t)mb * KB + kb) << 9) + lane * 16 + elem] = to_bf(src[idx]);
}

// B tile 32(K)x16(N) bf16: lane holds column n=lane%16, same K order as A: lane<16 K[0..7],K[16..23]; lane>=16 K[8..15],K[24..31]
__global__ __launch_bounds__(256) void packB_kernel(const float* __restrict__ src,
                                                    __hip_bfloat16* __restrict__ dst,
                                                    int K, int Nsrc, int NT, int ntOff) {
    size_t idx = (size_t)blockIdx.x * blockDim.x + threadIdx.x;
    size_t tot = (size_t)K * Nsrc;
    if (idx >= tot) return;
    int k = (int)(idx / Nsrc), n = (int)(idx % Nsrc);
    int kb = k >> 5, kr = k & 31;
    int lane = (n & 15) + (((kr >> 3) & 1) ? 16 : 0);
    int elem = (kr & 7) + ((kr >> 4) << 3);
    int nt   = (n >> 4) + ntOff;
    dst[(((size_t)kb * NT + nt) << 9) + lane * 16 + elem] = __float2bfloat16(src[idx]);
}

// ---------------- WMMA GEMM: one wave per 16 x (16*NW) output strip ----------------
// A tile loaded once per k-step, reused across NW column tiles -> 1+NW loads per NW WMMAs.
// EPI 0: out = C + bias (f32)
// EPI 1: out = resid + scale[0] * (C + bias) (f32)
// EPI 2: outb = gelu(C + bias) (bf16 row-major)
template <int EPI, int NW>
__global__ __launch_bounds__(256) void gemm_bf16(const __hip_bfloat16* __restrict__ Apk,
                                                 const __hip_bfloat16* __restrict__ Bpk,
                                                 const float* __restrict__ bias,
                                                 const float* __restrict__ resid,
                                                 const float* __restrict__ scalep,
                                                 float* __restrict__ out,
                                                 __hip_bfloat16* __restrict__ outb,
                                                 int Mtiles, int NT, int KB) {
    int gw   = (int)((blockIdx.x * 256u + threadIdx.x) >> 5);
    int lane = threadIdx.x & 31;
    int ng   = NT / NW;                       // n-groups per row strip
    if (gw >= Mtiles * ng) return;
    int mb  = gw / ng;
    int nt0 = (gw % ng) * NW;

    v8f acc[NW];
    #pragma unroll
    for (int t = 0; t < NW; ++t) acc[t] = (v8f){0.f, 0.f, 0.f, 0.f, 0.f, 0.f, 0.f, 0.f};

    const v16bf* A = (const v16bf*)Apk + (size_t)mb * KB * 32 + lane;
    const v16bf* B = (const v16bf*)Bpk + (size_t)nt0 * 32 + lane;
    for (int kb = 0; kb < KB; ++kb) {
        v16bf a = A[(size_t)kb * 32];
        #pragma unroll
        for (int t = 0; t < NW; ++t) {
            v16bf b = B[(size_t)(kb * NT + t) * 32];
            acc[t] = __builtin_amdgcn_wmma_f32_16x16x32_bf16(false, a, false, b,
                                                             (short)0, acc[t], false, false);
        }
    }

    int rowb = mb * 16 + ((lane < 16) ? 0 : 8);
    int ldc  = NT * 16;
    float sc = (EPI == 1) ? scalep[0] : 0.f;
    #pragma unroll
    for (int t = 0; t < NW; ++t) {
        int col  = (nt0 + t) * 16 + (lane & 15);
        float bv = bias[col];
        #pragma unroll
        for (int r = 0; r < 8; ++r) {
            float v  = acc[t][r] + bv;
            size_t o = (size_t)(rowb + r) * ldc + col;
            if (EPI == 0) {
                out[o] = v;
            } else if (EPI == 1) {
                out[o] = resid[o] + sc * v;
            } else {
                float gl = 0.5f * v * (1.0f + erff(v * 0.70710678118654752f));
                outb[o] = __float2bfloat16(gl);
            }
        }
    }
}

// ---------------- edge attention ----------------
__global__ __launch_bounds__(256) void init_seg_kernel(unsigned* __restrict__ mkey,
                                                       float* __restrict__ ssum) {
    int i = blockIdx.x * blockDim.x + threadIdx.x;
    if (i < NN * HH) { mkey[i] = 0x007FFFFFu /* fkey(-inf) */; ssum[i] = 0.f; }
}
__global__ __launch_bounds__(256) void zero_f32_kernel(float* __restrict__ p, int n) {
    int i = blockIdx.x * blockDim.x + threadIdx.x;
    if (i < n) p[i] = 0.f;
}

__global__ __launch_bounds__(256) void edge_scores_kernel(const float* __restrict__ qkv,
                                                          const float* __restrict__ ep,
                                                          const float* __restrict__ ew,
                                                          const int* __restrict__ ei,
                                                          float* __restrict__ scores,
                                                          unsigned* __restrict__ mkey) {
    int idx = blockIdx.x * blockDim.x + threadIdx.x;
    if (idx >= EE * HH) return;
    int e = idx >> 3, h = idx & 7;
    int src = ei[e], dst = ei[EE + e];
    const float4* qp  = (const float4*)(qkv + (size_t)dst * 768 + h * 32);
    const float4* kp  = (const float4*)(qkv + (size_t)src * 768 + 256 + h * 32);
    const float4* epp = (const float4*)(ep + (size_t)e * 32);
    float d = 0.f;
    #pragma unroll
    for (int j = 0; j < 8; ++j) {
        float4 q = qp[j], k = kp[j], p = epp[j];
        d += q.x * (k.x + p.x) + q.y * (k.y + p.y) + q.z * (k.z + p.z) + q.w * (k.w + p.w);
    }
    float sc = d * 0.17677669529663687f * ew[e];   // * DH^-0.5 * edge_weight
    scores[idx] = sc;
    atomicMax(&mkey[dst * HH + h], fkey(sc));
}

__global__ __launch_bounds__(256) void edge_exp_kernel(const int* __restrict__ ei,
                                                       float* __restrict__ scores,
                                                       const unsigned* __restrict__ mkey,
                                                       float* __restrict__ ssum) {
    int idx = blockIdx.x * blockDim.x + threadIdx.x;
    if (idx >= EE * HH) return;
    int e = idx >> 3, h = idx & 7;
    int dst = ei[EE + e];
    float m  = fkey_dec(mkey[dst * HH + h]);       // finite along any existing edge
    float ex = expf(scores[idx] - m);
    scores[idx] = ex;
    atomicAdd(&ssum[dst * HH + h], ex);
}

__global__ __launch_bounds__(256) void edge_agg_kernel(const int* __restrict__ ei,
                                                       const float* __restrict__ scores,
                                                       const float* __restrict__ ssum,
                                                       const float* __restrict__ qkv,
                                                       float* __restrict__ agg) {
    int idx = blockIdx.x * blockDim.x + threadIdx.x;
    if (idx >= EE * HH) return;
    int e = idx >> 3, h = idx & 7;
    int src = ei[e], dst = ei[EE + e];
    float coef = scores[idx] / ssum[dst * HH + h];
    const float4* vp = (const float4*)(qkv + (size_t)src * 768 + 512 + h * 32);
    float* ap = agg + (size_t)dst * DD + h * 32;
    #pragma unroll
    for (int j = 0; j < 8; ++j) {
        float4 v = vp[j];
        atomicAdd(ap + 4 * j + 0, coef * v.x);
        atomicAdd(ap + 4 * j + 1, coef * v.y);
        atomicAdd(ap + 4 * j + 2, coef * v.z);
        atomicAdd(ap + 4 * j + 3, coef * v.w);
    }
}

__global__ void concat_bias_kernel(const float* __restrict__ bq, const float* __restrict__ bk,
                                   const float* __restrict__ bv, float* __restrict__ o) {
    int i = blockIdx.x * blockDim.x + threadIdx.x;
    if (i < DD) { o[i] = bq[i]; o[DD + i] = bk[i]; o[2 * DD + i] = bv[i]; }
}

// ---------------- driver ----------------
static inline int blks(size_t n) { return (int)((n + 255) / 256); }
static inline int wblks(int waves) { return (waves + 7) / 8; }   // 8 waves per 256-thread block

extern "C" void kernel_launch(void* const* d_in, const int* in_sizes, int n_in,
                              void* d_out, int out_size, void* d_ws, size_t ws_size,
                              hipStream_t stream) {
    (void)in_sizes; (void)n_in; (void)out_size; (void)ws_size;
    const float* x    = (const float*)d_in[0];
    const float* ef   = (const float*)d_in[1];
    const float* ew   = (const float*)d_in[2];
    const int*   ei   = (const int*)d_in[3];
    const float* Wq   = (const float*)d_in[4];
    const float* bq   = (const float*)d_in[5];
    const float* Wk   = (const float*)d_in[6];
    const float* bk   = (const float*)d_in[7];
    const float* Wv   = (const float*)d_in[8];
    const float* bv   = (const float*)d_in[9];
    const float* We   = (const float*)d_in[10];
    const float* be   = (const float*)d_in[11];
    const float* Wo   = (const float*)d_in[12];
    const float* bo   = (const float*)d_in[13];
    const float* ln1g = (const float*)d_in[14];
    const float* ln1b = (const float*)d_in[15];
    const float* ln2g = (const float*)d_in[16];
    const float* ln2b = (const float*)d_in[17];
    const float* W1   = (const float*)d_in[18];
    const float* b1   = (const float*)d_in[19];
    const float* W2   = (const float*)d_in[20];
    const float* b2   = (const float*)d_in[21];
    const float* alpha = (const float*)d_in[22];
    const float* beta  = (const float*)d_in[23];
    float* out = (float*)d_out;

    // workspace carve (aliased by lifetime; ~187 MB total)
    char* w = (char*)d_ws;
    auto alloc = [&](size_t bytes) { void* p = (void*)w; w += (bytes + 255) & ~(size_t)255; return p; };
    float*          xn_x1 = (float*)alloc((size_t)NN * DD * 4);            // xn1, later x1
    __hip_bfloat16* apk   = (__hip_bfloat16*)alloc((size_t)NN * DD * 2);   // packed-A (xn1/agg/xn2)
    float*          qkv   = (float*)alloc((size_t)NN * 768 * 4);           // qkv; later xn2 + hb
    __hip_bfloat16* r4    = (__hip_bfloat16*)alloc((size_t)EE * EDD * 2);  // ef_pk, later h_pk
    float*          r5    = (float*)alloc((size_t)EE * DHH * 4);           // ep, later agg
    float*          scores = (float*)alloc((size_t)EE * HH * 4);
    unsigned*       mkey   = (unsigned*)alloc((size_t)NN * HH * 4);
    float*          ssum   = (float*)alloc((size_t)NN * HH * 4);
    __hip_bfloat16* Bqkv   = (__hip_bfloat16*)alloc((size_t)256 * 768 * 2);
    __hip_bfloat16* Bwe    = (__hip_bfloat16*)alloc((size_t)64 * 32 * 2);
    __hip_bfloat16* Bwo    = (__hip_bfloat16*)alloc((size_t)256 * 256 * 2);
    __hip_bfloat16* Bw1    = (__hip_bfloat16*)alloc((size_t)256 * 1024 * 2);
    __hip_bfloat16* Bw2    = (__hip_bfloat16*)alloc((size_t)1024 * 256 * 2);
    float*          bqkv   = (float*)alloc(768 * 4);

    float*          xn2 = qkv;                                          // after qkv dead
    __hip_bfloat16* hb  = (__hip_bfloat16*)(qkv + (size_t)NN * DD);     // bf16 [N,FF] row-major
    float*          agg = r5;                                           // after ep dead

    // weight / bias packing (tiny)
    packB_kernel<<<blks((size_t)256 * 256), 256, 0, stream>>>(Wq, Bqkv, 256, 256, 48, 0);
    packB_kernel<<<blks((size_t)256 * 256), 256, 0, stream>>>(Wk, Bqkv, 256, 256, 48, 16);
    packB_kernel<<<blks((size_t)256 * 256), 256, 0, stream>>>(Wv, Bqkv, 256, 256, 48, 32);
    packB_kernel<<<blks((size_t)64 * 32), 256, 0, stream>>>(We, Bwe, 64, 32, 2, 0);
    packB_kernel<<<blks((size_t)256 * 256), 256, 0, stream>>>(Wo, Bwo, 256, 256, 16, 0);
    packB_kernel<<<blks((size_t)256 * 1024), 256, 0, stream>>>(W1, Bw1, 256, 1024, 64, 0);
    packB_kernel<<<blks((size_t)1024 * 256), 256, 0, stream>>>(W2, Bw2, 1024, 256, 16, 0);
    concat_bias_kernel<<<1, 256, 0, stream>>>(bq, bk, bv, bqkv);

    // 1) LN1 + pack
    ln_kernel<<<(NN + 7) / 8, 256, 0, stream>>>(x, ln1g, ln1b, xn_x1, NN);
    packA_kernel<float><<<blks((size_t)NN * DD), 256, 0, stream>>>(xn_x1, apk, NN, DD);

    // 2) fused QKV GEMM: [20000,256] x [256,768], 16x64 strip per wave
    gemm_bf16<0, 4><<<wblks(1250 * 12), 256, 0, stream>>>(apk, Bqkv, bqkv, nullptr, nullptr,
                                                          qkv, nullptr, 1250, 48, 8);

    // 3) edge projection GEMM: [320000,64] x [64,32], 16x32 strip per wave
    packA_kernel<float><<<blks((size_t)EE * EDD), 256, 0, stream>>>(ef, r4, EE, EDD);
    gemm_bf16<0, 2><<<wblks(20000 * 1), 256, 0, stream>>>(r4, Bwe, be, nullptr, nullptr,
                                                          r5, nullptr, 20000, 2, 2);

    // 4) per-edge attention (segment softmax via atomics)
    init_seg_kernel<<<blks((size_t)NN * HH), 256, 0, stream>>>(mkey, ssum);
    edge_scores_kernel<<<blks((size_t)EE * HH), 256, 0, stream>>>(qkv, r5, ew, ei, scores, mkey);
    zero_f32_kernel<<<blks((size_t)NN * DD), 256, 0, stream>>>(agg, NN * DD);   // ep now dead
    edge_exp_kernel<<<blks((size_t)EE * HH), 256, 0, stream>>>(ei, scores, mkey, ssum);
    edge_agg_kernel<<<blks((size_t)EE * HH), 256, 0, stream>>>(ei, scores, ssum, qkv, agg);

    // 5) Wo GEMM with fused residual: x1 = x + alpha*(agg@Wo + bo)
    packA_kernel<float><<<blks((size_t)NN * DD), 256, 0, stream>>>(agg, apk, NN, DD);
    gemm_bf16<1, 4><<<wblks(1250 * 4), 256, 0, stream>>>(apk, Bwo, bo, x, alpha,
                                                         xn_x1, nullptr, 1250, 16, 8);

    // 6) LN2 + pack (qkv buffer now dead -> xn2 lives there)
    ln_kernel<<<(NN + 7) / 8, 256, 0, stream>>>(xn_x1, ln2g, ln2b, xn2, NN);
    packA_kernel<float><<<blks((size_t)NN * DD), 256, 0, stream>>>(xn2, apk, NN, DD);

    // 7) FFN up GEMM with fused exact GELU -> bf16 [N,1024]
    gemm_bf16<2, 4><<<wblks(1250 * 16), 256, 0, stream>>>(apk, Bw1, b1, nullptr, nullptr,
                                                          nullptr, hb, 1250, 64, 8);
    packA_kernel<__hip_bfloat16><<<blks((size_t)NN * FFF), 256, 0, stream>>>(hb, r4, NN, FFF);

    // 8) FFN down GEMM with fused residual: out = x1 + beta*(h@W2 + b2)
    gemm_bf16<1, 4><<<wblks(1250 * 4), 256, 0, stream>>>(r4, Bw2, b2, xn_x1, beta,
                                                         out, nullptr, 1250, 16, 32);
}
